// PC_37477884625141
// MI455X (gfx1250) — compile-verified, matrix operand order checked by reading, not run on hardware
//
#include <hip/hip_runtime.h>

// ---------------- constants (match reference) ----------------
constexpr int kB    = 4096;
constexpr int kOBS  = 784;
constexpr int kLAT  = 64;
constexpr int kHID  = 256;
constexpr int kSTEPS = 20;
constexpr int kOUT  = 784 + 64 + 64 + 784;   // 1696
constexpr float kLR   = 0.9f;
constexpr float kDAMP = 0.9f;                 // DAMP_ERR == DAMP_ACT
constexpr float kTDC  = 2.0f * 0.1f * 0.1f / (float)kLAT;  // 3.125e-4

typedef _Float16 v16h __attribute__((ext_vector_type(16)));
typedef _Float16 v8h  __attribute__((ext_vector_type(8)));
typedef float    v8f  __attribute__((ext_vector_type(8)));
typedef int      pc_v4i __attribute__((vector_size(16)));

// ---- CDNA5 async global->LDS copy (ASYNCcnt) with safe fallback ----
#if defined(__AMDGCN__) && \
    __has_builtin(__builtin_amdgcn_global_load_async_to_lds_b128) && \
    __has_builtin(__builtin_amdgcn_s_wait_asynccnt)
#define PC_ASYNC 1
#else
#define PC_ASYNC 0
#endif

#if PC_ASYNC
typedef __attribute__((address_space(1))) pc_v4i* pc_gp_t;  // global int4*
typedef __attribute__((address_space(3))) pc_v4i* pc_lp_t;  // LDS int4*
#endif

__device__ __forceinline__ void pc_cp16(_Float16* lds, const _Float16* g) {
#if PC_ASYNC
  __builtin_amdgcn_global_load_async_to_lds_b128(
      (pc_gp_t)(uintptr_t)g,
      (pc_lp_t)(uint32_t)(uintptr_t)lds,
      0, 0);
#else
  *reinterpret_cast<uint4*>(lds) = *reinterpret_cast<const uint4*>(g);
#endif
}
__device__ __forceinline__ void pc_async_wait() {
#if PC_ASYNC
  __builtin_amdgcn_s_wait_asynccnt(0);
#endif
}

// =====================================================================
// Über-GEMM (all-f16 operands, f32 WMMA accumulation):
//   acc = A[M,K] @ (TRANSB ? Bw[N,K]^T : Bw[K,N])
//   v = act( alpha*acc + b0*P0 + b1*P1 ) [ * (mask>0) ]
//   out32 (f32, ld32) and/or out16 (f16, ld16)
// Block = 128 threads (4 wave32 waves) -> one 64x64 C tile.
// Fast staging: 16B async copies straight into LDS (stride 40 halves,
// 16B-aligned), consumed as b128 fragment loads by v_wmma_f32_16x16x32_f16.
// =====================================================================
template <bool TRANSB>
__global__ __launch_bounds__(128) void pc_gemm16(
    const _Float16* __restrict__ A, int ldA,
    const _Float16* __restrict__ Bw,
    int M, int N, int K,
    float alpha,
    const _Float16* __restrict__ P0, float b0,
    const _Float16* __restrict__ P1, float b1,
    const _Float16* __restrict__ maskp,
    int act,
    float* __restrict__ out32, int ld32,
    _Float16* __restrict__ out16, int ld16) {
  __shared__ _Float16 As[64 * 40];
  __shared__ _Float16 Bs[64 * 40];

  const int tid  = threadIdx.x;
  const int wave = tid >> 5;
  const int lane = tid & 31;
  const int m0 = blockIdx.x * 64;
  const int n0 = blockIdx.y * 64;

  // fragment-build lane mapping (ISA 7.12.2, f16 16x32 A layout)
  const int halfsel = lane >> 4;
  const int r   = lane & 15;
  const int kb0 = halfsel ? 8 : 0;
  const int kb1 = halfsel ? 24 : 16;

  const bool fullM = (m0 + 64 <= M);
  const bool fullN = (n0 + 64 <= N);

  v8f acc[4] = {};

  for (int kb = 0; kb < K; kb += 32) {
    const bool fullK = (kb + 32 <= K);

    // ---- stage A tile [64 rows x 32 k] : 16B chunks, async ----
    if (fullM && fullK) {
#pragma unroll
      for (int i = 0; i < 2; ++i) {
        const int chunk = tid + i * 128;        // 256 chunks of 8 halves
        const int row = chunk >> 2;
        const int k8  = (chunk & 3) * 8;
        pc_cp16(&As[row * 40 + k8], A + (size_t)(m0 + row) * ldA + kb + k8);
      }
    } else {
#pragma unroll
      for (int i = 0; i < 16; ++i) {
        int idx = tid + i * 128;
        int row = idx >> 5, kk = idx & 31;
        int gr = m0 + row, gk = kb + kk;
        As[row * 40 + kk] =
            (gr < M && gk < K) ? A[(size_t)gr * ldA + gk] : (_Float16)0.0f;
      }
    }

    // ---- stage B tile [64 n x 32 k] (n-major, k-contiguous in LDS) ----
    if (TRANSB) {
      if (fullN && fullK) {
#pragma unroll
        for (int i = 0; i < 2; ++i) {
          const int chunk = tid + i * 128;
          const int n  = chunk >> 2;
          const int k8 = (chunk & 3) * 8;
          pc_cp16(&Bs[n * 40 + k8], Bw + (size_t)(n0 + n) * K + kb + k8);
        }
      } else {
#pragma unroll
        for (int i = 0; i < 16; ++i) {
          int idx = tid + i * 128;
          int n = idx >> 5, kk = idx & 31;
          int gn = n0 + n, gk = kb + kk;
          Bs[n * 40 + kk] =
              (gn < N && gk < K) ? Bw[(size_t)gn * K + gk] : (_Float16)0.0f;
        }
      }
    } else {
      if (fullN && fullK) {
#pragma unroll
        for (int i = 0; i < 2; ++i) {
          const int chunk = tid + i * 128;      // 32 k x 8 n-groups
          const int n8 = (chunk & 7) * 8;
          const int kk = chunk >> 3;
          const v8h v = *reinterpret_cast<const v8h*>(
              Bw + (size_t)(kb + kk) * N + n0 + n8);
#pragma unroll
          for (int j = 0; j < 8; ++j) Bs[(n8 + j) * 40 + kk] = v[j];
        }
      } else {
#pragma unroll
        for (int i = 0; i < 16; ++i) {
          int idx = tid + i * 128;
          int n = idx >> 5, kk = idx & 31;
          int gn = n0 + n, gk = kb + kk;
          Bs[n * 40 + kk] =
              (gn < N && gk < K) ? Bw[(size_t)gk * N + gn] : (_Float16)0.0f;
        }
      }
    }
    pc_async_wait();
    __syncthreads();

    v16h af;
    const _Float16* ar = &As[(wave * 16 + r) * 40];
#pragma unroll
    for (int i = 0; i < 8; ++i) { af[i] = ar[kb0 + i]; af[8 + i] = ar[kb1 + i]; }

#pragma unroll
    for (int t = 0; t < 4; ++t) {
      v16h bf;
      const _Float16* br = &Bs[(t * 16 + r) * 40];
#pragma unroll
      for (int i = 0; i < 8; ++i) { bf[i] = br[kb0 + i]; bf[8 + i] = br[kb1 + i]; }
      acc[t] = __builtin_amdgcn_wmma_f32_16x16x32_f16(
          false, af, false, bf, (short)0, acc[t], false, false);
    }
    __syncthreads();
  }

  // epilogue: C layout — vgpr r8 -> local row r8 + 8*halfsel, col = lane&15
#pragma unroll
  for (int t = 0; t < 4; ++t) {
#pragma unroll
    for (int r8 = 0; r8 < 8; ++r8) {
      int row = m0 + wave * 16 + r8 + 8 * halfsel;
      int col = n0 + t * 16 + (lane & 15);
      if (row < M && col < N) {
        float v = alpha * acc[t][r8];
        size_t e = (size_t)row * N + col;
        if (P0)    v += b0 * (float)P0[e];
        if (P1)    v += b1 * (float)P1[e];
        if (maskp) v = ((float)maskp[e] > 0.0f) ? v : 0.0f;
        if (act)   v = fmaxf(v, 0.0f);
        if (out32) out32[(size_t)row * ld32 + col] = v;
        if (out16) out16[(size_t)row * ld16 + col] = (_Float16)v;
      }
    }
  }
}

// =====================================================================
// Covariance: out[64,64] (f32) = (E^T E)/kB + damp*I ; E: [kB,64] f16.
// Single block, 4 waves; one shared tile serves both WMMA operands.
// =====================================================================
__global__ __launch_bounds__(128) void pc_cov16(const _Float16* __restrict__ E,
                                                float* __restrict__ outp,
                                                float damp) {
  __shared__ _Float16 Ts[64 * 40];
  const int tid  = threadIdx.x;
  const int wave = tid >> 5;
  const int lane = tid & 31;
  const int halfsel = lane >> 4;
  const int r   = lane & 15;
  const int kb0 = halfsel ? 8 : 0;
  const int kb1 = halfsel ? 24 : 16;

  v8f acc[4] = {};

  for (int kb = 0; kb < kB; kb += 32) {
#pragma unroll
    for (int i = 0; i < 2; ++i) {
      const int chunk = tid + i * 128;          // 32 k x 8 feat-groups
      const int f8 = (chunk & 7) * 8;
      const int kk = chunk >> 3;
      const v8h v = *reinterpret_cast<const v8h*>(E + (size_t)(kb + kk) * 64 + f8);
#pragma unroll
      for (int j = 0; j < 8; ++j) Ts[(f8 + j) * 40 + kk] = v[j];
    }
    __syncthreads();

    v16h af;
    const _Float16* ar = &Ts[(wave * 16 + r) * 40];
#pragma unroll
    for (int i = 0; i < 8; ++i) { af[i] = ar[kb0 + i]; af[8 + i] = ar[kb1 + i]; }
#pragma unroll
    for (int t = 0; t < 4; ++t) {
      v16h bf;
      const _Float16* br = &Ts[(t * 16 + r) * 40];
#pragma unroll
      for (int i = 0; i < 8; ++i) { bf[i] = br[kb0 + i]; bf[8 + i] = br[kb1 + i]; }
      acc[t] = __builtin_amdgcn_wmma_f32_16x16x32_f16(
          false, af, false, bf, (short)0, acc[t], false, false);
    }
    __syncthreads();
  }

  const float inv_b = 1.0f / (float)kB;
#pragma unroll
  for (int t = 0; t < 4; ++t)
#pragma unroll
    for (int r8 = 0; r8 < 8; ++r8) {
      int row = wave * 16 + r8 + 8 * halfsel;
      int col = t * 16 + (lane & 15);
      outp[row * 64 + col] = acc[t][r8] * inv_b + ((row == col) ? damp : 0.0f);
    }
}

// =====================================================================
// 64x64 Gauss-Jordan inverse in f32 LDS; emits f16 for WMMA consumers.
// =====================================================================
__global__ __launch_bounds__(256) void pc_inv64(const float* __restrict__ G,
                                                _Float16* __restrict__ Inv) {
  __shared__ float aug[64][130];
  const int tid = threadIdx.x;
  const int r   = tid >> 2;     // 0..63
  const int seg = tid & 3;      // 4 threads per row, 32 cols each

#pragma unroll 4
  for (int i = 0; i < 32; ++i) {
    int c = seg * 32 + i;
    aug[r][c] = (c < 64) ? G[r * 64 + c] : ((c - 64 == r) ? 1.0f : 0.0f);
  }

  for (int k = 0; k < 64; ++k) {
    __syncthreads();
    float pinv = 1.0f / aug[k][k];
    __syncthreads();
    if (r == k)
      for (int i = 0; i < 32; ++i) aug[k][seg * 32 + i] *= pinv;
    __syncthreads();
    float f = aug[r][k];
    __syncthreads();
    if (r != k)
      for (int i = 0; i < 32; ++i) aug[r][seg * 32 + i] -= f * aug[k][seg * 32 + i];
  }
  __syncthreads();

#pragma unroll 4
  for (int i = 0; i < 32; ++i) {
    int c = seg * 32 + i;
    if (c >= 64) Inv[r * 64 + (c - 64)] = (_Float16)aug[r][c];
  }
}

__global__ void pc_fill16(_Float16* __restrict__ p, float v, int n) {
  int i = blockIdx.x * blockDim.x + threadIdx.x;
  if (i < n) p[i] = (_Float16)v;
}

__global__ void pc_cvt16(const float* __restrict__ s, _Float16* __restrict__ d,
                         int n) {
  int i = blockIdx.x * blockDim.x + threadIdx.x;
  if (i < n) d[i] = (_Float16)s[i];
}

// =====================================================================
// Host orchestration
// =====================================================================
extern "C" void kernel_launch(void* const* d_in, const int* in_sizes, int n_in,
                              void* d_out, int out_size, void* d_ws, size_t ws_size,
                              hipStream_t stream) {
  (void)in_sizes; (void)n_in; (void)out_size; (void)ws_size;

  const float* xf = (const float*)d_in[0];
  // d_in[1] = prior (unused: prior0 = prior*0 + 1e-5)
  const float* wf[3][3] = {
    { (const float*)d_in[2], (const float*)d_in[3], (const float*)d_in[4] },   // dec1
    { (const float*)d_in[5], (const float*)d_in[6], (const float*)d_in[7] },   // dec2
    { (const float*)d_in[8], (const float*)d_in[9], (const float*)d_in[10] },  // dec3
  };
  float* out = (float*)d_out;
  const int xd[3] = { kOBS, kLAT, kLAT };

  // workspace carve-up
  char* base = (char*)d_ws;
  size_t off = 0;
  auto allocH = [&](size_t n) {
    _Float16* q = (_Float16*)(base + off);
    off = (off + n * sizeof(_Float16) + 15) & ~(size_t)15;
    return q;
  };
  auto allocF = [&](size_t n) {
    float* q = (float*)(base + off);
    off = (off + n * sizeof(float) + 15) & ~(size_t)15;
    return q;
  };

  _Float16* xh = allocH((size_t)kB * kOBS);
  _Float16* wh[3][3];
  for (int L = 0; L < 3; ++L) {
    wh[L][0] = allocH((size_t)kHID * kLAT);
    wh[L][1] = allocH((size_t)kHID * kHID);
    wh[L][2] = allocH((size_t)xd[L] * kHID);
  }
  _Float16* bufH0  = allocH((size_t)kB * kHID);
  _Float16* bufH1  = allocH((size_t)kB * kHID);
  _Float16* bufGH1 = allocH((size_t)kB * kHID);
  _Float16* bufBig = allocH((size_t)kB * kOBS);   // g_pred, then g_h0
  _Float16* bufG   = allocH((size_t)kB * kLAT);   // g, then g2
  _Float16* mu[3]  = { allocH((size_t)kB * kLAT), allocH((size_t)kB * kLAT),
                       allocH((size_t)kB * kLAT) };
  _Float16* lat2 = allocH((size_t)kB * kLAT);
  _Float16* lat3 = allocH((size_t)kB * kLAT);
  _Float16* pri  = allocH((size_t)kB * kLAT);
  _Float16* bufP2 = allocH((size_t)kB * kLAT);
  _Float16* Ei = allocH(64 * 64);
  _Float16* Si = allocH(64 * 64);
  _Float16* Q  = allocH(64 * 64);
  float*    covT = allocF(64 * 64);

  auto cvt = [&](const float* s, _Float16* d, int n) {
    pc_cvt16<<<(n + 255) / 256, 256, 0, stream>>>(s, d, n);
  };

  auto gemm = [&](const _Float16* A, int ldA, const _Float16* Bw,
                  int M, int N, int K, bool tB, float alpha,
                  const _Float16* P0, float b0, const _Float16* P1, float b1,
                  const _Float16* mask, int act,
                  float* o32, int ld32, _Float16* o16, int ld16) {
    dim3 g((M + 63) / 64, (N + 63) / 64);
    if (tB)
      pc_gemm16<true><<<g, 128, 0, stream>>>(A, ldA, Bw, M, N, K, alpha,
                                             P0, b0, P1, b1, mask, act,
                                             o32, ld32, o16, ld16);
    else
      pc_gemm16<false><<<g, 128, 0, stream>>>(A, ldA, Bw, M, N, K, alpha,
                                              P0, b0, P1, b1, mask, act,
                                              o32, ld32, o16, ld16);
  };

  // decoder: h0=relu(z@w0^T); h1=relu(h0@w1^T); out = alpha*(h1@wm^T)+b0*P0
  auto dec_fwd = [&](int L, const _Float16* z, int ldz,
                     float* o32, int ld32, _Float16* o16, int ld16,
                     float alpha, const _Float16* P0, float b0) {
    gemm(z, ldz, wh[L][0], kB, kHID, kLAT, true, 1.f, nullptr, 0.f, nullptr, 0.f,
         nullptr, 1, nullptr, 0, bufH0, kHID);
    gemm(bufH0, kHID, wh[L][1], kB, kHID, kHID, true, 1.f, nullptr, 0.f,
         nullptr, 0.f, nullptr, 1, nullptr, 0, bufH1, kHID);
    gemm(bufH1, kHID, wh[L][2], kB, xd[L], kHID, true, alpha, P0, b0,
         nullptr, 0.f, nullptr, 0, o32, ld32, o16, ld16);
  };

  // ---- one-time f16 conversions (weights + x) ----
  cvt(xf, xh, kB * kOBS);
  for (int L = 0; L < 3; ++L) {
    cvt(wf[L][0], wh[L][0], kHID * kLAT);
    cvt(wf[L][1], wh[L][1], kHID * kHID);
    cvt(wf[L][2], wh[L][2], xd[L] * kHID);
  }

  // ---- init: prior0 = 1e-5; lat3 = dec3(p); lat2 = dec2(lat3) ----
  pc_fill16<<<(kB * kLAT + 255) / 256, 256, 0, stream>>>(pri, 1e-5f, kB * kLAT);
  dec_fwd(2, pri,  kLAT, nullptr, 0, lat3, kLAT, 1.f, nullptr, 0.f);
  dec_fwd(1, lat3, kLAT, nullptr, 0, lat2, kLAT, 1.f, nullptr, 0.f);
  const size_t latBytes = (size_t)kB * kLAT * sizeof(_Float16);
  (void)hipMemcpyAsync(mu[0], lat2, latBytes, hipMemcpyDeviceToDevice, stream);
  (void)hipMemcpyAsync(mu[1], lat3, latBytes, hipMemcpyDeviceToDevice, stream);
  (void)hipMemcpyAsync(mu[2], pri,  latBytes, hipMemcpyDeviceToDevice, stream);

  // ---- inference loop ----
  const _Float16* tgt[3] = { xh, mu[0], mu[1] };
  const _Float16* td[3]  = { lat2, lat3, pri };

  for (int s = 0; s < kSTEPS; ++s) {
    for (int L = 0; L < 3; ++L) {
      const float a = 2.0f / (float)xd[L];
      // forward + fused g_pred = (2/xd)*(pred - target)
      dec_fwd(L, mu[L], kLAT, nullptr, 0, bufBig, xd[L], a, tgt[L], -a);
      // g_h1 = (g_pred @ wm) * (h1 > 0)
      gemm(bufBig, xd[L], wh[L][2], kB, kHID, xd[L], false, 1.f,
           nullptr, 0.f, nullptr, 0.f, bufH1, 0, nullptr, 0, bufGH1, kHID);
      // g_h0 = (g_h1 @ w1) * (h0 > 0)   (reuse bufBig)
      gemm(bufGH1, kHID, wh[L][1], kB, kHID, kHID, false, 1.f,
           nullptr, 0.f, nullptr, 0.f, bufH0, 0, nullptr, 0, bufBig, kHID);
      // g = g_h0 @ w0 + c*(mu - td)
      gemm(bufBig, kHID, wh[L][0], kB, kLAT, kHID, false, 1.f,
           mu[L], kTDC, td[L], -kTDC, nullptr, 0, nullptr, 0, bufG, kLAT);
      // E = inv(cov(g)+damp I); S = inv(cov(mu)+damp I)
      pc_cov16<<<1, 128, 0, stream>>>(bufG, covT, kDAMP);
      pc_inv64<<<1, 256, 0, stream>>>(covT, Ei);
      pc_cov16<<<1, 128, 0, stream>>>(mu[L], covT, kDAMP);
      pc_inv64<<<1, 256, 0, stream>>>(covT, Si);
      // (S@E)^T = E@S (symmetric):  Q = Ei @ Si
      gemm(Ei, 64, Si, 64, 64, 64, false, 1.f, nullptr, 0.f, nullptr, 0.f,
           nullptr, 0, nullptr, 0, Q, 64);
      // mu = mu - LR * (g @ Q)
      gemm(bufG, kLAT, Q, kB, kLAT, kLAT, false, -kLR, mu[L], 1.f,
           nullptr, 0.f, nullptr, 0, nullptr, 0, mu[L], kLAT);
    }
  }

  // ---- outputs: [pp0 | pp1 | pp2 | pred_global] into [kB, 1696] ----
  dec_fwd(0, mu[0], kLAT, out + 0,   kOUT, nullptr, 0, 1.f, nullptr, 0.f); // pp0
  dec_fwd(1, mu[1], kLAT, out + 784, kOUT, nullptr, 0, 1.f, nullptr, 0.f); // pp1
  dec_fwd(2, mu[2], kLAT, out + 848, kOUT, bufP2, kLAT, 1.f, nullptr, 0.f);// pp2
  dec_fwd(1, bufP2, kLAT, nullptr, 0, bufG, kLAT, 1.f, nullptr, 0.f);      // g2
  dec_fwd(0, bufG,  kLAT, out + 912, kOUT, nullptr, 0, 1.f, nullptr, 0.f); // pred_global
}
